// HAN_23948737642663
// MI455X (gfx1250) — matrix operand order repaired and checked by reading, not compiled, not run on hardware
//
#include <hip/hip_runtime.h>
#include <hip/hip_bf16.h>
#include <math.h>

// ---------------------------------------------------------------------------
// CDNA5 WMMA types
// ---------------------------------------------------------------------------
typedef __attribute__((ext_vector_type(16))) __bf16 v16bf;
typedef __attribute__((ext_vector_type(8)))  float  v8f;
typedef __attribute__((ext_vector_type(8)))  unsigned int v8u;

__device__ __forceinline__ unsigned short f2bf(float f) {
  union { float f; unsigned u; } v; v.f = f;
  unsigned r = v.u + 0x7FFFu + ((v.u >> 16) & 1u);   // round-to-nearest-even
  return (unsigned short)(r >> 16);
}

// Branch-free activations on the CDNA5 transcendental pipe (v_exp_f32 / v_rcp_f32).
__device__ __forceinline__ float sigmoidf_(float x) {
  return __builtin_amdgcn_rcpf(1.0f + __expf(-x));
}
__device__ __forceinline__ float tanhf_(float x) {
  // tanh(x) = 2/(1+e^(-2x)) - 1 ; saturates correctly at +-1 via inf/0 of exp
  return 2.0f * __builtin_amdgcn_rcpf(1.0f + __expf(-2.0f * x)) - 1.0f;
}

__device__ __forceinline__ uint4 pack8(const float4 a, const float4 b) {
  union { unsigned short s[8]; uint4 u; } v;
  v.s[0] = f2bf(a.x); v.s[1] = f2bf(a.y); v.s[2] = f2bf(a.z); v.s[3] = f2bf(a.w);
  v.s[4] = f2bf(b.x); v.s[5] = f2bf(b.y); v.s[6] = f2bf(b.z); v.s[7] = f2bf(b.w);
  return v.u;
}

// A fragment: 16x32 bf16 tile from LDS, row-major [16, ld] (ld padded, 8-even)
// lane m=lane&15; elems 0..15 = K kb+khalf+0..15 (khalf = 16*(lane>>4))
__device__ __forceinline__ v16bf load_a_frag(const unsigned short* A, int ld, int kb, int lane) {
  int m = lane & 15;
  int khalf = (lane >> 4) << 4;
  const unsigned short* p = A + m * ld + kb + khalf;
  union { uint4 u[2]; v16bf b; } v;
  v.u[0] = *(const uint4*)p;
  v.u[1] = *(const uint4*)(p + 8);
  return v.b;
}

// B fragment from pre-packed lane-order storage: [frag][lane][16 bf16]
// (one contiguous 32-byte load per lane -> 2x global_load_b128)
__device__ __forceinline__ v16bf load_b_packed(const unsigned short* Bp, int frag, int lane) {
  union { v8u u; v16bf b; } v;
  v.u = *(const v8u*)(Bp + (((long)frag * 32 + lane) << 4));
  return v.b;
}

#define WMMA_BF16(A, B, C) \
  __builtin_amdgcn_wmma_f32_16x16x32_bf16(false, (A), false, (B), (short)0, (C), false, false)

// ---------------------------------------------------------------------------
// Problem constants
// ---------------------------------------------------------------------------
static constexpr int Bc = 32, Sc = 32, Tc = 64;
static constexpr int Ec = 128;
static constexpr int Vc = 50000, NCc = 8;

// ---------------------------------------------------------------------------
// Prep kernels
// ---------------------------------------------------------------------------
__global__ void HAN_cvt_bf16(const float* __restrict__ src, unsigned short* __restrict__ dst, int n) {
  int i = blockIdx.x * 256 + threadIdx.x;
  if (i < n) dst[i] = f2bf(src[i]);
}

// Pack logical B = src^T (src [R,C] f32; B is [K=C, N=R]) into WMMA lane order:
// frag = nt*(C/32)+kt ; dst[(frag*32+lane)*16+e] = B[kt*32+(lane>=16)*16+e][nt*16+(lane&15)]
__global__ void HAN_pack_bT(const float* __restrict__ src, unsigned short* __restrict__ dst,
                            int R, int C) {
  int i = blockIdx.x * 256 + threadIdx.x;
  if (i >= R * C) return;
  int e    = i & 15;
  int lane = (i >> 4) & 31;
  int rest = i >> 9;
  int KT   = C >> 5;
  int kt   = rest % KT;
  int nt   = rest / KT;
  int n = nt * 16 + (lane & 15);
  int k = kt * 32 + ((lane >> 4) << 4) + e;
  dst[i] = f2bf(src[n * C + k]);
}

__global__ void HAN_lens(const int* __restrict__ x, int* __restrict__ wlens, int* __restrict__ slens) {
  int gid = blockIdx.x * 256 + threadIdx.x;
  if (gid < Bc * Sc) {
    const int* row = x + gid * Tc;
    int len = 1;
    for (int t = Tc - 1; t >= 0; --t) { if (row[t] != 0) { len = t + 1; break; } }
    wlens[gid] = len;
  }
  if (gid < Bc) {
    const int* doc = x + gid * Sc * Tc;
    int len = 1;
    for (int s = Sc - 1; s >= 0; --s) {
      bool nz = false;
      for (int t = 0; t < Tc; ++t) if (doc[s * Tc + t] != 0) { nz = true; break; }
      if (nz) { len = s + 1; break; }
    }
    slens[gid] = len;
  }
}

__global__ void HAN_maxlens(const int* __restrict__ wlens, const int* __restrict__ slens,
                            int* __restrict__ maxout) {
  __shared__ int sm[256];
  int tid = threadIdx.x;
  int m = 0;
  for (int i = tid; i < Bc * Sc; i += 256) m = max(m, wlens[i]);
  sm[tid] = m; __syncthreads();
  for (int s = 128; s > 0; s >>= 1) { if (tid < s) sm[tid] = max(sm[tid], sm[tid + s]); __syncthreads(); }
  if (tid == 0) maxout[0] = sm[0];
  __syncthreads();
  sm[tid] = (tid < Bc) ? slens[tid] : 0; __syncthreads();
  for (int s = 128; s > 0; s >>= 1) { if (tid < s) sm[tid] = max(sm[tid], sm[tid + s]); __syncthreads(); }
  if (tid == 0) maxout[1] = sm[0];
}

// ---------------------------------------------------------------------------
// Bidirectional GRU: each workgroup owns 16 independent rows for one direction
// and loops all T steps with h resident in LDS. 8 waves; wave w produces gate
// columns w*16..w*16+15 for r (p=0), z (p=1), n (p=2) via WMMA bf16.
// ---------------------------------------------------------------------------
template <int DIN, bool GATHER>
__global__ __launch_bounds__(256) void HAN_gru(
    const int* __restrict__ tokens, const float* __restrict__ xin,
    const unsigned short* __restrict__ emb16,
    const unsigned short* __restrict__ wihP_f, const unsigned short* __restrict__ whhP_f,
    const float* __restrict__ bih_f, const float* __restrict__ bhh_f,
    const unsigned short* __restrict__ wihP_b, const unsigned short* __restrict__ whhP_b,
    const float* __restrict__ bih_b, const float* __restrict__ bhh_b,
    const int* __restrict__ lens, float* __restrict__ out, int T) {
  constexpr int KTI = DIN / 32;          // k-tiles of Wih
  constexpr int LDX = DIN + 8;           // padded LDS row (ushorts)
  __shared__ float hS[16][128];
  __shared__ float outS[16][128];
  __shared__ unsigned short Xs[16][LDX];
  __shared__ unsigned short Hs[16][136];
  __shared__ int lenS[16];

  const int tid = threadIdx.x;
  const int lane = tid & 31;
  const int w = tid >> 5;
  const int rowbase = blockIdx.x * 16;
  const int dir = blockIdx.y;

  const unsigned short* wihP = dir ? wihP_b : wihP_f;
  const unsigned short* whhP = dir ? whhP_b : whhP_f;
  const float* bih = dir ? bih_b : bih_f;
  const float* bhh = dir ? bhh_b : bhh_f;

  if (tid < 16) lenS[tid] = lens[rowbase + tid];
  for (int i = tid; i < 16 * 128; i += 256) (&hS[0][0])[i] = 0.0f;
  __syncthreads();

  const int n  = lane & 15;
  const int mb = (lane >> 4) << 3;
  const int c  = w * 16 + n;                 // gate column within [0,128)
  const float br  = bih[c] + bhh[c];
  const float bz  = bih[128 + c] + bhh[128 + c];
  const float bxn = bih[256 + c];
  const float bhn = bhh[256 + c];

  for (int step = 0; step < T; ++step) {
    const int tt = dir ? (T - 1 - step) : step;

    // ---- stage X tile (bf16) : 8-element (16B) chunks per thread ----
    if constexpr (GATHER) {
      for (int i = tid; i < 16 * (DIN / 8); i += 256) {
        int m = i / (DIN / 8), ch = i % (DIN / 8);
        int tok = tokens[(rowbase + m) * T + tt];
        *(uint4*)&Xs[m][ch * 8] = *(const uint4*)&emb16[(long)tok * Ec + ch * 8];
      }
    } else {
      for (int i = tid; i < 16 * (DIN / 8); i += 256) {
        int m = i / (DIN / 8), ch = i % (DIN / 8);
        const float* s = &xin[((long)(rowbase + m) * T + tt) * DIN + ch * 8];
        *(uint4*)&Xs[m][ch * 8] = pack8(*(const float4*)s, *(const float4*)(s + 4));
      }
    }
    // ---- stage H tile (bf16) ----
    {
      int m = tid >> 4, ch = tid & 15;
      const float* s = &hS[m][ch * 8];
      *(uint4*)&Hs[m][ch * 8] = pack8(*(const float4*)s, *(const float4*)(s + 4));
    }
    __syncthreads();

    const v8f z8 = {0.f, 0.f, 0.f, 0.f, 0.f, 0.f, 0.f, 0.f};
    v8f aX[3] = {z8, z8, z8};
    v8f aH[3] = {z8, z8, z8};

#pragma unroll
    for (int kt = 0; kt < KTI; ++kt) {
      v16bf a = load_a_frag(&Xs[0][0], LDX, kt * 32, lane);
#pragma unroll
      for (int p = 0; p < 3; ++p) {
        v16bf b = load_b_packed(wihP, (p * 8 + w) * KTI + kt, lane);
        aX[p] = WMMA_BF16(a, b, aX[p]);
      }
    }
#pragma unroll
    for (int kt = 0; kt < 4; ++kt) {
      v16bf a = load_a_frag(&Hs[0][0], 136, kt * 32, lane);
#pragma unroll
      for (int p = 0; p < 3; ++p) {
        v16bf b = load_b_packed(whhP, (p * 8 + w) * 4 + kt, lane);
        aH[p] = WMMA_BF16(a, b, aH[p]);
      }
    }

#pragma unroll
    for (int i = 0; i < 8; ++i) {
      int m = mb + i;
      float r  = sigmoidf_(aX[0][i] + aH[0][i] + br);
      float zg = sigmoidf_(aX[1][i] + aH[1][i] + bz);
      float ng = tanhf_(aX[2][i] + bxn + r * (aH[2][i] + bhn));
      float hold = hS[m][c];
      float hnew = (1.0f - zg) * ng + zg * hold;
      bool valid = tt < lenS[m];
      hS[m][c]   = valid ? hnew : hold;
      outS[m][c] = valid ? hnew : 0.0f;
    }
    __syncthreads();

    // ---- coalesced store of the 16x128 output tile ----
    {
      int m = tid >> 4, cb = (tid & 15) * 8;
      float* d = &out[((long)(rowbase + m) * T + tt) * 256 + dir * 128 + cb];
      *(float4*)d       = *(const float4*)&outS[m][cb];
      *(float4*)(d + 4) = *(const float4*)&outS[m][cb + 4];
    }
  }
}

// ---------------------------------------------------------------------------
// Attention scores: score = tanh(seq @ projW^T + pb) . ctx   (seq width 256)
// 16 flat rows per WG; wave w -> proj columns w*16..w*16+15 via WMMA.
// ---------------------------------------------------------------------------
__global__ __launch_bounds__(256) void HAN_attn_scores(
    const float* __restrict__ seq, const unsigned short* __restrict__ projP,
    const float* __restrict__ pb, const float* __restrict__ ctx,
    float* __restrict__ scores) {
  __shared__ unsigned short As[16][264];
  __shared__ float red[16][128];
  const int tid = threadIdx.x, lane = tid & 31, w = tid >> 5;
  const long rowbase = (long)blockIdx.x * 16;

  for (int i = tid; i < 16 * 32; i += 256) {     // 8-elem chunks
    int m = i >> 5, ch = i & 31;
    const float* s = &seq[(rowbase + m) * 256 + ch * 8];
    *(uint4*)&As[m][ch * 8] = pack8(*(const float4*)s, *(const float4*)(s + 4));
  }
  __syncthreads();

  v8f acc = {0.f, 0.f, 0.f, 0.f, 0.f, 0.f, 0.f, 0.f};
#pragma unroll
  for (int kt = 0; kt < 8; ++kt) {
    v16bf a = load_a_frag(&As[0][0], 264, kt * 32, lane);
    v16bf b = load_b_packed(projP, w * 8 + kt, lane);
    acc = WMMA_BF16(a, b, acc);
  }
  const int n = lane & 15, mbq = (lane >> 4) << 3, c = w * 16 + n;
  const float bpc = pb[c], cc = ctx[c];
#pragma unroll
  for (int i = 0; i < 8; ++i) red[mbq + i][c] = tanhf_(acc[i] + bpc) * cc;
  __syncthreads();
  if (tid < 16) {
    float s = 0.f;
    for (int j = 0; j < 128; ++j) s += red[tid][j];
    scores[rowbase + tid] = s;
  }
}

// ---------------------------------------------------------------------------
// Softmax over t < Tmax (global max of lens, per reference) + weighted pool.
// ---------------------------------------------------------------------------
__global__ __launch_bounds__(256) void HAN_softmax_pool(
    const float* __restrict__ scores, const float* __restrict__ seq,
    const int* __restrict__ maxlen, float* __restrict__ pooled, int T) {
  __shared__ float p[64];
  __shared__ float ssum;
  const int tid = threadIdx.x;
  const int row = blockIdx.x;
  const int Tmax = *maxlen;
  if (tid < T) p[tid] = (tid < Tmax) ? scores[row * T + tid] : -INFINITY;
  __syncthreads();
  if (tid == 0) {
    float m = -INFINITY;
    for (int t = 0; t < T; ++t) m = fmaxf(m, p[t]);
    float s = 0.f;
    for (int t = 0; t < T; ++t) { float e = __expf(p[t] - m); p[t] = e; s += e; }
    ssum = s;
  }
  __syncthreads();
  const float inv = 1.0f / ssum;
  float acc = 0.f;
  for (int t = 0; t < T; ++t) acc += p[t] * seq[((long)row * T + t) * 256 + tid];
  pooled[row * 256 + tid] = acc * inv;
}

__global__ void HAN_logits(const float* __restrict__ doc, const float* __restrict__ fcW,
                           const float* __restrict__ fcb, float* __restrict__ out) {
  int tid = threadIdx.x;            // 256 = 32 rows x 8 classes
  int r = tid >> 3, k = tid & 7;
  float s = fcb[k];
  for (int j = 0; j < 256; ++j) s += doc[r * 256 + j] * fcW[k * 256 + j];
  out[r * NCc + k] = s;
}

// ---------------------------------------------------------------------------
// Workspace layout (bytes, 256-aligned)
// ---------------------------------------------------------------------------
static constexpr size_t ALGN(size_t x) { return (x + 255) & ~size_t(255); }
static constexpr size_t OFF_WOUT   = 0;                                           // [1024,64,256] f32
static constexpr size_t OFF_SENT   = OFF_WOUT   + ALGN(1024ull * 64 * 256 * 4);   // [1024,256] f32
static constexpr size_t OFF_SOUT   = OFF_SENT   + ALGN(1024ull * 256 * 4);        // [32,32,256] f32
static constexpr size_t OFF_DOC    = OFF_SOUT   + ALGN(32ull * 32 * 256 * 4);     // [32,256] f32
static constexpr size_t OFF_WSC    = OFF_DOC    + ALGN(32ull * 256 * 4);          // [1024,64] f32
static constexpr size_t OFF_SSC    = OFF_WSC    + ALGN(1024ull * 64 * 4);         // [32,32] f32
static constexpr size_t OFF_WLEN   = OFF_SSC    + ALGN(32ull * 32 * 4);           // [1024] i32
static constexpr size_t OFF_SLEN   = OFF_WLEN   + ALGN(1024ull * 4);              // [32] i32
static constexpr size_t OFF_MAXL   = OFF_SLEN   + ALGN(32ull * 4);                // [2] i32
static constexpr size_t OFF_EMB16  = OFF_MAXL   + ALGN(2ull * 4);                 // [50000,128] bf16
static constexpr size_t OFF_WFIHP  = OFF_EMB16  + ALGN(50000ull * 128 * 2);       // packed [384,128]
static constexpr size_t OFF_WFHHP  = OFF_WFIHP  + ALGN(128ull * 384 * 2);
static constexpr size_t OFF_WBIHP  = OFF_WFHHP  + ALGN(128ull * 384 * 2);
static constexpr size_t OFF_WBHHP  = OFF_WBIHP  + ALGN(128ull * 384 * 2);
static constexpr size_t OFF_SFIHP  = OFF_WBHHP  + ALGN(128ull * 384 * 2);         // packed [384,256]
static constexpr size_t OFF_SFHHP  = OFF_SFIHP  + ALGN(256ull * 384 * 2);
static constexpr size_t OFF_SBIHP  = OFF_SFHHP  + ALGN(128ull * 384 * 2);
static constexpr size_t OFF_SBHHP  = OFF_SBIHP  + ALGN(256ull * 384 * 2);
static constexpr size_t OFF_WPROJP = OFF_SBHHP  + ALGN(128ull * 384 * 2);         // packed [128,256]
static constexpr size_t OFF_SPROJP = OFF_WPROJP + ALGN(256ull * 128 * 2);

extern "C" void kernel_launch(void* const* d_in, const int* in_sizes, int n_in,
                              void* d_out, int out_size, void* d_ws, size_t ws_size,
                              hipStream_t stream) {
  (void)in_sizes; (void)n_in; (void)out_size; (void)ws_size;
  const int*   x       = (const int*)  d_in[0];
  const float* emb     = (const float*)d_in[1];
  const float* wf_Wih  = (const float*)d_in[2],  *wf_Whh = (const float*)d_in[3];
  const float* wf_bih  = (const float*)d_in[4],  *wf_bhh = (const float*)d_in[5];
  const float* wb_Wih  = (const float*)d_in[6],  *wb_Whh = (const float*)d_in[7];
  const float* wb_bih  = (const float*)d_in[8],  *wb_bhh = (const float*)d_in[9];
  const float* sf_Wih  = (const float*)d_in[10], *sf_Whh = (const float*)d_in[11];
  const float* sf_bih  = (const float*)d_in[12], *sf_bhh = (const float*)d_in[13];
  const float* sb_Wih  = (const float*)d_in[14], *sb_Whh = (const float*)d_in[15];
  const float* sb_bih  = (const float*)d_in[16], *sb_bhh = (const float*)d_in[17];
  const float* wproj_W = (const float*)d_in[18], *wproj_b = (const float*)d_in[19];
  const float* wctx    = (const float*)d_in[20];
  const float* sproj_W = (const float*)d_in[21], *sproj_b = (const float*)d_in[22];
  const float* sctx    = (const float*)d_in[23];
  const float* fc_W    = (const float*)d_in[24], *fc_b    = (const float*)d_in[25];
  float* out = (float*)d_out;

  char* ws = (char*)d_ws;
  float* WOUT  = (float*)(ws + OFF_WOUT);
  float* SENT  = (float*)(ws + OFF_SENT);
  float* SOUT  = (float*)(ws + OFF_SOUT);
  float* DOC   = (float*)(ws + OFF_DOC);
  float* WSC   = (float*)(ws + OFF_WSC);
  float* SSC   = (float*)(ws + OFF_SSC);
  int*   WLEN  = (int*)  (ws + OFF_WLEN);
  int*   SLEN  = (int*)  (ws + OFF_SLEN);
  int*   MAXL  = (int*)  (ws + OFF_MAXL);
  unsigned short* EMB16  = (unsigned short*)(ws + OFF_EMB16);
  unsigned short* WFIHP  = (unsigned short*)(ws + OFF_WFIHP);
  unsigned short* WFHHP  = (unsigned short*)(ws + OFF_WFHHP);
  unsigned short* WBIHP  = (unsigned short*)(ws + OFF_WBIHP);
  unsigned short* WBHHP  = (unsigned short*)(ws + OFF_WBHHP);
  unsigned short* SFIHP  = (unsigned short*)(ws + OFF_SFIHP);
  unsigned short* SFHHP  = (unsigned short*)(ws + OFF_SFHHP);
  unsigned short* SBIHP  = (unsigned short*)(ws + OFF_SBIHP);
  unsigned short* SBHHP  = (unsigned short*)(ws + OFF_SBHHP);
  unsigned short* WPROJP = (unsigned short*)(ws + OFF_WPROJP);
  unsigned short* SPROJP = (unsigned short*)(ws + OFF_SPROJP);

  // --- prep: bf16 conversion + WMMA-lane-order weight packing ---
  HAN_cvt_bf16<<<(Vc * Ec + 255) / 256, 256, 0, stream>>>(emb, EMB16, Vc * Ec);
  HAN_pack_bT<<<(384 * 128) / 256, 256, 0, stream>>>(wf_Wih, WFIHP, 384, 128);
  HAN_pack_bT<<<(384 * 128) / 256, 256, 0, stream>>>(wf_Whh, WFHHP, 384, 128);
  HAN_pack_bT<<<(384 * 128) / 256, 256, 0, stream>>>(wb_Wih, WBIHP, 384, 128);
  HAN_pack_bT<<<(384 * 128) / 256, 256, 0, stream>>>(wb_Whh, WBHHP, 384, 128);
  HAN_pack_bT<<<(384 * 256) / 256, 256, 0, stream>>>(sf_Wih, SFIHP, 384, 256);
  HAN_pack_bT<<<(384 * 128) / 256, 256, 0, stream>>>(sf_Whh, SFHHP, 384, 128);
  HAN_pack_bT<<<(384 * 256) / 256, 256, 0, stream>>>(sb_Wih, SBIHP, 384, 256);
  HAN_pack_bT<<<(384 * 128) / 256, 256, 0, stream>>>(sb_Whh, SBHHP, 384, 128);
  HAN_pack_bT<<<(128 * 256) / 256, 256, 0, stream>>>(wproj_W, WPROJP, 128, 256);
  HAN_pack_bT<<<(128 * 256) / 256, 256, 0, stream>>>(sproj_W, SPROJP, 128, 256);

  HAN_lens<<<4, 256, 0, stream>>>(x, WLEN, SLEN);
  HAN_maxlens<<<1, 256, 0, stream>>>(WLEN, SLEN, MAXL);

  // --- word-level BiGRU: 1024 seqs, T=64, fused gather + WMMA recurrence ---
  HAN_gru<128, true><<<dim3(64, 2), 256, 0, stream>>>(
      x, nullptr, EMB16, WFIHP, WFHHP, wf_bih, wf_bhh,
      WBIHP, WBHHP, wb_bih, wb_bhh, WLEN, WOUT, Tc);

  // --- word attention -> sentence vectors ---
  HAN_attn_scores<<<(1024 * 64) / 16, 256, 0, stream>>>(WOUT, WPROJP, wproj_b, wctx, WSC);
  HAN_softmax_pool<<<1024, 256, 0, stream>>>(WSC, WOUT, MAXL + 0, SENT, Tc);

  // --- sentence-level BiGRU: 32 docs, T=32, DIN=256 ---
  HAN_gru<256, false><<<dim3(2, 2), 256, 0, stream>>>(
      nullptr, SENT, nullptr, SFIHP, SFHHP, sf_bih, sf_bhh,
      SBIHP, SBHHP, sb_bih, sb_bhh, SLEN, SOUT, Sc);

  // --- sentence attention -> doc vectors ---
  HAN_attn_scores<<<(32 * 32) / 16, 256, 0, stream>>>(SOUT, SPROJP, sproj_b, sctx, SSC);
  HAN_softmax_pool<<<32, 256, 0, stream>>>(SSC, SOUT, MAXL + 1, DOC, Sc);

  // --- classifier ---
  HAN_logits<<<1, 256, 0, stream>>>(DOC, fc_W, fc_b, out);
}
